// LIAFLSTMCell_8332236554272
// MI455X (gfx1250) — matrix operand order repaired and checked by reading, not compiled, not run on hardware
//
#include <hip/hip_runtime.h>

// ---------------------------------------------------------------------------
// Types for CDNA5 WMMA
// ---------------------------------------------------------------------------
typedef __attribute__((ext_vector_type(16))) __bf16 v16bf;
typedef __attribute__((ext_vector_type(8)))  __bf16 v8bf;
typedef __attribute__((ext_vector_type(8)))  float  v8f;

union Frag { v16bf v; v8bf h[2]; };

// Problem constants (match reference: B=512, T=64, DIN=1024, H=1024)
#define M_DIM  32768        // B*T
#define K_DIM  1024         // DIN
#define N_DIM  4096         // 4 gates * H
#define H_DIM  1024
#define T_DIM  64

#define TILE_M 128
#define TILE_N 128
#define TILE_K 32
#define LDSS   40           // padded LDS row stride (elements): conflict-free
#define NKIT   (K_DIM / TILE_K)   // 32
#define MATSZ  (TILE_M * LDSS)    // 5120 elements per matrix tile

__device__ __forceinline__ unsigned short f2bf(float f) {
  unsigned int u = __float_as_uint(f);
  u += 0x7FFFu + ((u >> 16) & 1u);   // round-to-nearest-even
  return (unsigned short)(u >> 16);
}

// CDNA5 async global->LDS copy (GVS mode): no data VGPRs, tracked by ASYNCcnt.
// LDS address comes from lds_addr; global address is the SGPR base plus voff
__device__ __forceinline__ void async_ld_b128(unsigned lds_addr, unsigned voff,
                                              const unsigned short* sbase) {
  asm volatile("global_load_async_to_lds_b128 %0, %1, %2"
               :: "v"(lds_addr), "v"(voff), "s"(sbase)
               : "memory");
}
__device__ __forceinline__ void wait_async0() {
  asm volatile("s_wait_asynccnt 0" ::: "memory");
}

// ---------------------------------------------------------------------------
// Kernel 1: convert x (fp32) -> bf16, 4 elements/thread
// ---------------------------------------------------------------------------
__global__ __launch_bounds__(256) void cvt_x_kernel(
    const float4* __restrict__ x, unsigned long long* __restrict__ xb) {
  int i = blockIdx.x * 256 + threadIdx.x;      // 8388608 float4 chunks
  float4 v = x[i];
  unsigned long long p = (unsigned long long)f2bf(v.x)
      | ((unsigned long long)f2bf(v.y) << 16)
      | ((unsigned long long)f2bf(v.z) << 32)
      | ((unsigned long long)f2bf(v.w) << 48);
  xb[i] = p;
}

// ---------------------------------------------------------------------------
// Kernel 2: pack Wi/Wf/Wg/Wo (each H x DIN fp32) into one (4096 x 1024) bf16
// ---------------------------------------------------------------------------
__global__ __launch_bounds__(256) void pack_w_kernel(
    const float* __restrict__ Wi, const float* __restrict__ Wf,
    const float* __restrict__ Wg, const float* __restrict__ Wo,
    unsigned short* __restrict__ Wb) {
  int i = blockIdx.x * 256 + threadIdx.x;      // 0 .. 4194303
  int gate = i >> 20;                          // H*DIN = 1<<20
  int rem  = i & ((1 << 20) - 1);
  const float* src = (gate == 0) ? Wi : (gate == 1) ? Wf : (gate == 2) ? Wg : Wo;
  Wb[i] = f2bf(src[rem]);
}

// ---------------------------------------------------------------------------
// Kernel 3: pack biases into bcat[4096]
// ---------------------------------------------------------------------------
__global__ __launch_bounds__(256) void pack_b_kernel(
    const float* __restrict__ bi, const float* __restrict__ bf_,
    const float* __restrict__ bg, const float* __restrict__ bo,
    float* __restrict__ bcat) {
  int i = blockIdx.x * 256 + threadIdx.x;      // 0 .. 4095
  int gate = i >> 10;
  int h = i & (H_DIM - 1);
  const float* src = (gate == 0) ? bi : (gate == 1) ? bf_ : (gate == 2) ? bg : bo;
  bcat[i] = src[h];
}

// ---------------------------------------------------------------------------
// Kernel 4: GEMM  z[M,4096] = A[M,K](bf16) * B[4096,K](bf16)^T + bias
// 256 threads = 8 wave32 waves; block tile 128x128; wave tile 64x32 (4x2 WMMA)
// Double-buffered LDS filled by GLOBAL_LOAD_ASYNC_TO_LDS_B128 (no staging
// VGPRs, no spills); one barrier per K-step; fully unrolled so accumulators
// stay in place (D==C) with no v_mov traffic.
// ---------------------------------------------------------------------------
__global__ __launch_bounds__(256) void gemm_bf16_wmma(
    const unsigned short* __restrict__ A,    // M x K row-major
    const unsigned short* __restrict__ B,    // N x K row-major (weights)
    const float* __restrict__ bias,          // N
    float* __restrict__ C)                   // M x N row-major
{
  __shared__ __align__(16) unsigned short lds[2 * 2 * MATSZ];  // 40 KB

  const int tid   = threadIdx.x;
  const int lane  = tid & 31;
  const int wv    = tid >> 5;
  const int lhalf = lane >> 4;      // g: 0 or 1 (lane group)
  const int lmod  = lane & 15;

  const long bm = (long)blockIdx.y * TILE_M;
  const long bn = (long)blockIdx.x * TILE_N;
  const int waveM = (wv >> 2) * 64;
  const int waveN = (wv & 3) * 32;

  const unsigned short* gA = A + bm * K_DIM;
  const unsigned short* gB = B + bn * K_DIM;

  // async-copy chunking: 512 x 16B chunks per 128x32 tile, 2 chunks/thread.
  // Per-lane offsets are FIXED for the whole kernel; the K advance is folded
  // into the uniform SGPR base pointer (scalar constant adds after unroll).
  const unsigned ldsbase = (unsigned)(unsigned long long)(const void*)&lds[0];
  unsigned gvo[2];          // per-lane global byte offset within the panel
  unsigned la[2][2], lb[2][2];  // LDS byte addrs: [buffer][chunk] for A / B
#pragma unroll
  for (int i = 0; i < 2; ++i) {
    int c = i * 256 + tid;
    int crow = c >> 2;             // 0..127
    int ccol = (c & 3) * 8;        // element offset 0/8/16/24
    gvo[i]   = (unsigned)((crow * K_DIM + ccol) * 2);
    la[0][i] = ldsbase + (unsigned)((crow * LDSS + ccol) * 2);
    lb[0][i] = la[0][i] + MATSZ * 2;
    la[1][i] = la[0][i] + 2 * MATSZ * 2;
    lb[1][i] = lb[0][i] + 2 * MATSZ * 2;
  }

  v8f acc[4][2];
#pragma unroll
  for (int mt = 0; mt < 4; ++mt)
#pragma unroll
    for (int nt = 0; nt < 2; ++nt)
      acc[mt][nt] = (v8f){0.f, 0.f, 0.f, 0.f, 0.f, 0.f, 0.f, 0.f};

  // prologue: async-fill buffer 0 with K-chunk 0
#pragma unroll
  for (int i = 0; i < 2; ++i) {
    async_ld_b128(la[0][i], gvo[i], gA);
    async_ld_b128(lb[0][i], gvo[i], gB);
  }
  wait_async0();
  __syncthreads();

#pragma unroll
  for (int kt = 0; kt < NKIT; ++kt) {
    // issue async DMA for the next K-chunk into the other buffer; it overlaps
    // with the WMMA work below and is fenced by s_wait_asynccnt + barrier.
    if (kt + 1 < NKIT) {
      const int nxt = (kt + 1) & 1;
      const unsigned short* gAn = gA + (kt + 1) * TILE_K;  // uniform SGPR add
      const unsigned short* gBn = gB + (kt + 1) * TILE_K;
#pragma unroll
      for (int i = 0; i < 2; ++i) {
        async_ld_b128(la[nxt][i], gvo[i], gAn);
        async_ld_b128(lb[nxt][i], gvo[i], gBn);
      }
    }

    const unsigned short* sA = lds + (kt & 1) * 2 * MATSZ;
    const unsigned short* sB = sA + MATSZ;

    // B fragments: lane = column n, positions p=0..15 -> K = g*16 + p
    Frag bfr[2];
#pragma unroll
    for (int nt = 0; nt < 2; ++nt) {
      const unsigned short* p = sB + (waveN + nt * 16 + lmod) * LDSS + lhalf * 16;
      bfr[nt].h[0] = *(const v8bf*)p;
      bfr[nt].h[1] = *(const v8bf*)(p + 8);
    }
    // A fragments: lane = row m, positions p=0..15 -> K = g*16 + p
#pragma unroll
    for (int mt = 0; mt < 4; ++mt) {
      Frag af;
      const unsigned short* p = sA + (waveM + mt * 16 + lmod) * LDSS + lhalf * 16;
      af.h[0] = *(const v8bf*)p;
      af.h[1] = *(const v8bf*)(p + 8);
#pragma unroll
      for (int nt = 0; nt < 2; ++nt) {
        acc[mt][nt] = __builtin_amdgcn_wmma_f32_16x16x32_bf16(
            false, af.v, false, bfr[nt].v, (short)0, acc[mt][nt], false, false);
      }
    }

    // ensure this wave's async fills landed, then rendezvous all waves
    wait_async0();
    __syncthreads();
  }

  // store: C/D layout — VGPR r: lanes0-15 M=r, lanes16-31 M=r+8; N = lane%16
#pragma unroll
  for (int mt = 0; mt < 4; ++mt) {
#pragma unroll
    for (int nt = 0; nt < 2; ++nt) {
      const long col  = bn + waveN + nt * 16 + lmod;
      const float bv  = bias[col];
      const long row0 = bm + waveM + mt * 16 + lhalf * 8;
      float* op = C + row0 * N_DIM + col;
#pragma unroll
      for (int r = 0; r < 8; ++r)
        op[(long)r * N_DIM] = acc[mt][nt][r] + bv;
    }
  }
}

// ---------------------------------------------------------------------------
// Kernel 5: recurrent LIAF scan over T. One thread per (b,h).
// ---------------------------------------------------------------------------
__global__ __launch_bounds__(256) void scan_kernel(
    const float* __restrict__ z,   // (B*T, 4096): [zi | zf | zg | zo]
    float* __restrict__ out)       // (B, T, H)
{
  const int gid = blockIdx.x * 256 + threadIdx.x;  // 0 .. 524287
  const int h = gid & (H_DIM - 1);
  const int b = gid >> 10;

  const float SC = 1.0507009873554805f;   // SELU scale
  const float AL = 1.6732632423543772f;   // SELU alpha
  const float TH = 0.25f, DEC = 0.3f;

  float vi = 0.f, vf = 0.f, vg = 0.f, vo = 0.f, c = 0.f;
  const float* zp = z + (size_t)b * T_DIM * N_DIM + h;
  float* op = out + (size_t)b * T_DIM * H_DIM + h;

  for (int t = 0; t < T_DIM; ++t) {
    const float zi  = zp[0];
    const float zf_ = zp[H_DIM];
    const float zg  = zp[2 * H_DIM];
    const float zo  = zp[3 * H_DIM];
    vi += zi; vf += zf_; vg += zg; vo += zo;

    const float i = fmaxf(vi, 0.f);
    const float f = fmaxf(vf, 0.f);
    const float o = fmaxf(vo, 0.f);
    const float g = (vg > 0.f) ? SC * vg : SC * AL * (__expf(vg) - 1.f);
    c = c * f + i * g;
    const float sc = (c > 0.f) ? SC * c : SC * AL * (__expf(c) - 1.f);
    op[0] = sc * o;

    vi = (vi > TH) ? 0.f : DEC * vi;
    vf = (vf > TH) ? 0.f : DEC * vf;
    vg = (vg > TH) ? 0.f : DEC * vg;
    vo = (vo > TH) ? 0.f : DEC * vo;

    zp += N_DIM;
    op += H_DIM;
  }
}

// ---------------------------------------------------------------------------
// Launch
// ---------------------------------------------------------------------------
extern "C" void kernel_launch(void* const* d_in, const int* in_sizes, int n_in,
                              void* d_out, int out_size, void* d_ws, size_t ws_size,
                              hipStream_t stream) {
  (void)in_sizes; (void)n_in; (void)out_size; (void)ws_size;

  const float* x   = (const float*)d_in[0];
  const float* Wi  = (const float*)d_in[1];
  const float* bi  = (const float*)d_in[2];
  const float* Wf  = (const float*)d_in[3];
  const float* bf_ = (const float*)d_in[4];
  const float* Wg  = (const float*)d_in[5];
  const float* bg  = (const float*)d_in[6];
  const float* Wo  = (const float*)d_in[7];
  const float* bo  = (const float*)d_in[8];
  float* out = (float*)d_out;

  // workspace layout
  char* ws = (char*)d_ws;
  const size_t XB_BYTES = (size_t)M_DIM * K_DIM * 2;          // 64 MiB
  const size_t WB_BYTES = (size_t)N_DIM * K_DIM * 2;          // 8 MiB
  const size_t BC_BYTES = (size_t)N_DIM * 4;                  // 16 KiB
  unsigned short* xb   = (unsigned short*)ws;
  unsigned short* Wb   = (unsigned short*)(ws + XB_BYTES);
  float*          bcat = (float*)(ws + XB_BYTES + WB_BYTES);
  float*          z    = (float*)(ws + XB_BYTES + WB_BYTES + BC_BYTES);

  // 1) convert x -> bf16  (33554432 elems, 4/thread)
  cvt_x_kernel<<<32768, 256, 0, stream>>>((const float4*)x,
                                          (unsigned long long*)xb);
  // 2) pack weights -> bf16 (4194304 elems)
  pack_w_kernel<<<16384, 256, 0, stream>>>(Wi, Wf, Wg, Wo, Wb);
  // 3) pack biases
  pack_b_kernel<<<16, 256, 0, stream>>>(bi, bf_, bg, bo, bcat);
  // 4) fused 4-gate GEMM with WMMA: grid (N/128, M/128) = (32, 256)
  dim3 ggrid(N_DIM / TILE_N, M_DIM / TILE_M);
  gemm_bf16_wmma<<<ggrid, 256, 0, stream>>>(xb, Wb, bcat, z);
  // 5) recurrent scan: 524288 threads
  scan_kernel<<<(M_DIM * H_DIM / T_DIM) / 256, 256, 0, stream>>>(z, out);
}
